// Self_Attention_1348619731323
// MI455X (gfx1250) — matrix operand order repaired and checked by reading, not compiled, not run on hardware
//
#include <hip/hip_runtime.h>
#include <hip/hip_bf16.h>

// ---------------------------------------------------------------------------
// Self-attention for MI455X (gfx1250, wave32, WMMA).
//
//   q,k,v = x @ W[0..2]            (M=16384, N=1024, K=1024 each)
//   S     = (q @ k^T) / sqrt(2048) per batch (2048x2048, K=1024)
//   P     = softmax_rows(S)
//   out   = P @ v                  (per batch 2048x1024, K=2048)
//
// ~240 GFLOP vs ~140MB mandatory HBM traffic -> compute bound -> f16 WMMA
// (V_WMMA_F32_16X16X32_F16, f32 accumulate). Scores (67MB f16) are
// materialized: the 192MB L2 holds them, so S/P never touches HBM.
//
// All three GEMMs use *direct-from-global* WMMA fragments (no LDS, no
// barriers): per CDNA5 ISA the f16 A fragment is two contiguous
// 16B runs of a row (lane L and L+16 read adjacent 16B of the same row ->
// full 64B cachelines), and the B fragment uses the same two 16B runs. This
// requires every operand to be f16 and memory-contiguous along K, which two
// tiny one-time prep kernels guarantee:
//   - cvt_x:    x fp32 -> x16 f16              (contiguous along K=d)
//   - trans_w:  W fp32 [3,d,o] -> Wt f16 [3,o,d] (contiguous along K=d)
// and the QKV epilogue writes V transposed (Vt[b][o][s], contiguous along
// K=s) — which is *cheaper* than the row-major store, since the 8 acc values
// per lane are 8 consecutive M => one 16B v8h store.
// ---------------------------------------------------------------------------

typedef __attribute__((ext_vector_type(16))) _Float16 v16h;
typedef __attribute__((ext_vector_type(8)))  _Float16 v8h;
typedef __attribute__((ext_vector_type(8)))  float    v8f;

#define B_DIM      8
#define S_DIM      2048
#define QKV_ELEMS  ((size_t)16384 * 1024)      // per Q/K/V buffer, halves
#define X16_OFF    ((size_t)0)                 // 16777216 halves
#define WT_OFF     ((size_t)16777216)          // 3145728 halves
#define QKV_OFF    ((size_t)19922944)          // 3*16777216 halves
#define S_OFF      ((size_t)70254592)          // 33554432 halves
#define SCALE_ATT  0.022097086912079608f       // 1/sqrt(2048)

#define WMMA_F16(a, b, c) \
  __builtin_amdgcn_wmma_f32_16x16x32_f16(false, (a), false, (b), (short)0, (c), false, false)

// Direct-from-global A fragment (row-major f16; caller adds lane K-offset):
//   lane<16 : m=lane,    halves[0..7]=K0..7,  halves[8..15]=K16..23
//   lane>=16: m=lane-16, halves[0..7]=K8..15, halves[8..15]=K24..31
__device__ __forceinline__ v16h loadAfrag(const _Float16* rowBase) {
  v16h a;
  *(v8h*)&a        = *(const v8h*)(rowBase);
  *(((v8h*)&a) + 1) = *(const v8h*)(rowBase + 16);
  return a;
}

// ===========================================================================
// Prep 1: x fp32 -> f16, elementwise.  16.7M elements, pure bandwidth.
// ===========================================================================
__global__ __launch_bounds__(256) void cvt_x(const float* __restrict__ X,
                                             _Float16* __restrict__ X16) {
  const size_t i = ((size_t)blockIdx.x * 256 + threadIdx.x) * 8;
  const float4 a = *(const float4*)(X + i);
  const float4 b = *(const float4*)(X + i + 4);
  v8h h;
  h[0] = (_Float16)a.x; h[1] = (_Float16)a.y;
  h[2] = (_Float16)a.z; h[3] = (_Float16)a.w;
  h[4] = (_Float16)b.x; h[5] = (_Float16)b.y;
  h[6] = (_Float16)b.z; h[7] = (_Float16)b.w;
  *(v8h*)(X16 + i) = h;
}

// ===========================================================================
// Prep 2: W fp32 [3,d,o] -> Wt f16 [3,o,d] via 32x32 LDS tile transpose.
// ===========================================================================
__global__ __launch_bounds__(256) void trans_w(const float* __restrict__ W,
                                               _Float16* __restrict__ Wt) {
  __shared__ _Float16 t[32][33];   // +1 pad: bank-conflict-free transpose
  const int which = blockIdx.z;
  const int d0 = blockIdx.x * 32, o0 = blockIdx.y * 32;
  const int c = threadIdx.x & 31, r4 = (threadIdx.x >> 5) * 4;
#pragma unroll
  for (int j = 0; j < 4; ++j) {
    const int r = r4 + j;
    t[r][c] = (_Float16)W[(size_t)which * 1048576 + (size_t)(d0 + r) * 1024 + o0 + c];
  }
  __syncthreads();
#pragma unroll
  for (int j = 0; j < 4; ++j) {
    const int r = r4 + j;
    Wt[((size_t)which * 1024 + o0 + r) * 1024 + d0 + c] = t[c][r];
  }
}

// ===========================================================================
// Unified direct-fragment GEMM. 256 threads = 8 waves (2Mx4N), wave tile
// 64x32 = 4x2 WMMA tiles, K-step 32. No LDS, no barriers.
//   MODE 0: qkv    A=x16 [16384,1024], B=Wt [3072,1024]   -> Q,K f16 + Vt f16
//   MODE 1: scores A=Q   [b,2048,1024], B=K  [b,2048,1024] -> S f16 * scale
//   MODE 2: av     A=P   [b,2048,2048], B=Vt [b,1024,2048] -> out fp32
// ===========================================================================
template <int MODE>
__global__ __launch_bounds__(256) void gemm_direct(const _Float16* __restrict__ A,
                                                   const _Float16* __restrict__ Bm,
                                                   void* __restrict__ OutP) {
  constexpr int K   = (MODE == 2) ? 2048 : 1024;
  constexpr int LDA = (MODE == 2) ? 2048 : 1024;
  constexpr int LDB = (MODE == 2) ? 2048 : 1024;

  const int tid  = threadIdx.x;
  const int lane = tid & 31;
  const int wave = tid >> 5;
  const int wm   = wave >> 2, wn = wave & 3;
  const int row0 = blockIdx.y * 128;
  const int col0 = blockIdx.x * 128;
  const int b    = blockIdx.z;

  const int mn    = lane & 15;
  const int aKoff = (lane >> 4) << 3;    // +8 halves for upper half-wave
  const int bKoff = (lane >> 4) << 3;    // +8 halves for upper half-wave

  const size_t aRow0 = (MODE == 0) ? 0 : (size_t)b * 2048;
  const size_t bRow0 = (MODE == 0) ? 0
                     : (MODE == 1) ? (size_t)b * 2048
                                   : (size_t)b * 1024;

  const _Float16* aBase[4];
  const _Float16* bBase[2];
#pragma unroll
  for (int i = 0; i < 4; ++i)
    aBase[i] = A + (aRow0 + row0 + (wm * 4 + i) * 16 + mn) * LDA + aKoff;
#pragma unroll
  for (int j = 0; j < 2; ++j)
    bBase[j] = Bm + (bRow0 + col0 + (wn * 2 + j) * 16 + mn) * LDB + bKoff;

  v8f acc[4][2] = {};

  for (int k0 = 0; k0 < K; k0 += 32) {
    v16h bfrag[2];
#pragma unroll
    for (int j = 0; j < 2; ++j) {
      bfrag[j] = loadAfrag(bBase[j] + k0);
      __builtin_prefetch(bBase[j] + k0 + 32, 0, 1);   // speculative: safe OOB
    }
#pragma unroll
    for (int i = 0; i < 4; ++i) {
      const v16h afrag = loadAfrag(aBase[i] + k0);
      __builtin_prefetch(aBase[i] + k0 + 32, 0, 1);
#pragma unroll
      for (int j = 0; j < 2; ++j) acc[i][j] = WMMA_F16(afrag, bfrag[j], acc[i][j]);
    }
  }

  // epilogue: C layout vgpr r -> M = r + (lane>=16?8:0), N = lane&15
#pragma unroll
  for (int i = 0; i < 4; ++i)
#pragma unroll
    for (int j = 0; j < 2; ++j) {
      const int mB = row0 + (wm * 4 + i) * 16 + ((lane >= 16) ? 8 : 0);
      const int n  = (wn * 2 + j) * 16 + mn;

      if constexpr (MODE == 0) {
        _Float16* OutH = (_Float16*)OutP;
        const int which = col0 >> 10;
        const int o     = (col0 & 1023) + n;
        if (which == 2) {
          // V transposed: Vt[b][o][s]; 8 acc values = 8 consecutive s
          v8h hv;
#pragma unroll
          for (int r = 0; r < 8; ++r) hv[r] = (_Float16)acc[i][j][r];
          const int bb = mB >> 11, s = mB & 2047;
          *(v8h*)&OutH[(size_t)2 * QKV_ELEMS + ((size_t)bb * 1024 + o) * 2048 + s] = hv;
        } else {
#pragma unroll
          for (int r = 0; r < 8; ++r)
            OutH[(size_t)which * QKV_ELEMS + (size_t)(mB + r) * 1024 + o] =
                (_Float16)acc[i][j][r];
        }
      } else if constexpr (MODE == 1) {
        _Float16* S = (_Float16*)OutP;
#pragma unroll
        for (int r = 0; r < 8; ++r)
          S[((size_t)b * 2048 + mB + r) * 2048 + col0 + n] =
              (_Float16)(acc[i][j][r] * SCALE_ATT);
      } else {
        float* Out = (float*)OutP;
#pragma unroll
        for (int r = 0; r < 8; ++r)
          Out[((size_t)b * 2048 + mB + r) * 1024 + col0 + n] = acc[i][j][r];
      }
    }
}

// ---------------------------------------------------------------------------
// Row softmax over 2048 f16 elements, in place. One block per (b, q) row.
// ---------------------------------------------------------------------------
__global__ __launch_bounds__(256) void softmax_rows(_Float16* __restrict__ s) {
  _Float16* p = s + (size_t)blockIdx.x * 2048;
  const int t = threadIdx.x;
  __shared__ float red[256];

  float f[8];
  float mx = -1e30f;
#pragma unroll
  for (int i = 0; i < 8; ++i) {
    f[i] = (float)p[t + i * 256];
    mx = fmaxf(mx, f[i]);
  }
  red[t] = mx;
  __syncthreads();
  for (int w = 128; w > 0; w >>= 1) {
    if (t < w) red[t] = fmaxf(red[t], red[t + w]);
    __syncthreads();
  }
  mx = red[0];
  __syncthreads();

  float sum = 0.f;
#pragma unroll
  for (int i = 0; i < 8; ++i) {
    f[i] = __expf(f[i] - mx);
    sum += f[i];
  }
  red[t] = sum;
  __syncthreads();
  for (int w = 128; w > 0; w >>= 1) {
    if (t < w) red[t] += red[t + w];
    __syncthreads();
  }
  const float inv = 1.0f / red[0];
#pragma unroll
  for (int i = 0; i < 8; ++i) p[t + i * 256] = (_Float16)(f[i] * inv);
}

// ---------------------------------------------------------------------------
extern "C" void kernel_launch(void* const* d_in, const int* in_sizes, int n_in,
                              void* d_out, int out_size, void* d_ws,
                              size_t ws_size, hipStream_t stream) {
  (void)in_sizes; (void)n_in; (void)out_size; (void)ws_size;

  const float* x  = (const float*)d_in[0];   // [8,2048,1024] fp32
  const float* wk = (const float*)d_in[1];   // [3,1024,1024] fp32
  float* out = (float*)d_out;                // [8,2048,1024] fp32

  // ws (halves): x16 32MB | Wt 6MB | Q 32MB | K 32MB | Vt 32MB | S 67MB
  _Float16* ws  = (_Float16*)d_ws;
  _Float16* X16 = ws + X16_OFF;
  _Float16* Wt  = ws + WT_OFF;
  _Float16* QKV = ws + QKV_OFF;              // Q | K | Vt
  _Float16* Q   = QKV;
  _Float16* Kh  = QKV + QKV_ELEMS;
  _Float16* Vt  = QKV + 2 * QKV_ELEMS;
  _Float16* S   = ws + S_OFF;

  const dim3 blk(256);
  cvt_x         <<<dim3(8192),       blk, 0, stream>>>(x, X16);
  trans_w       <<<dim3(32, 32, 3),  blk, 0, stream>>>(wk, Wt);
  gemm_direct<0><<<dim3(24, 128, 1), blk, 0, stream>>>(X16, Wt, QKV);
  gemm_direct<1><<<dim3(16, 16, 8),  blk, 0, stream>>>(Q, Kh, S);
  softmax_rows  <<<dim3(B_DIM * S_DIM), blk, 0, stream>>>(S);
  gemm_direct<2><<<dim3(8, 16, 8),   blk, 0, stream>>>(S, Vt, out);
}
